// SpatialSetEncoder_80711025426842
// MI455X (gfx1250) — matrix operand order repaired and checked by reading, not compiled, not run on hardware
//
#include <hip/hip_runtime.h>
#include <hip/hip_bf16.h>
#include <math.h>

// ---------------------------------------------------------------------------
// SpatialSetEncoder forward for MI455X (gfx1250, wave32).
// All MLP GEMMs run on v_wmma_f32_16x16x32_f16 (f16 A/B, f32 accumulate).
// Weights are packed TRANSPOSED (N x Kp) so both A and B fragments are
// pairs of 16B K-spans in LDS (k 8h.. and 16+8h..) -> ds_load_b128 pairs (no scalar gathers).
// Every GEMM has M%128==0, N%128==0, K%32==0 (K zero-padded: 6->32, 7->32,
// 1027->1056), so the hot loop has no bounds checks. 8 waves/block, each
// wave computes a 32x64 sub-tile (2x4 WMMA accumulators).
// ---------------------------------------------------------------------------

typedef _Float16 half8 __attribute__((ext_vector_type(8)));
typedef _Float16 v16h  __attribute__((ext_vector_type(16)));
typedef float    v8f   __attribute__((ext_vector_type(8)));

#define NPTS    16384
#define KNN_K   8
#define DM      256
#define HID     512
#define BASE_IN 259
#define XDIM    1027
#define XPAD    1056

#define BM 128
#define BN 128
#define BK 32
#define LDS_STRIDE 40      // halfs (32 + 8 pad) -> 80B rows, 16B aligned

#define EPI_SILU  1
#define EPI_RESID 2

// ---------------------------------------------------------------------------
__device__ __forceinline__ float block_reduce_sum(float v, float* sbuf) {
  const int tid = threadIdx.x;
  sbuf[tid] = v;
  __syncthreads();
  for (int s = 128; s > 0; s >>= 1) {
    if (tid < s) sbuf[tid] += sbuf[tid + s];
    __syncthreads();
  }
  const float r = sbuf[0];
  __syncthreads();
  return r;
}

// ---------------------------------------------------------------------------
// Pack f32 weight (K x N row-major) to f16 TRANSPOSED (N x Kp row-major),
// zero-padding K up to Kp. One-time cost; makes GEMM B-tiles identical in
// shape to A-tiles (contiguous along K).
__global__ void pack_weight_kernel(const float* __restrict__ w,
                                   _Float16* __restrict__ outT,
                                   int K, int N, int Kp) {
  const int idx = blockIdx.x * blockDim.x + threadIdx.x;
  if (idx >= Kp * N) return;
  const int n = idx / Kp, k = idx - n * Kp;
  outT[idx] = (k < K) ? (_Float16)w[k * N + n] : (_Float16)0.0f;
}

// ---------------------------------------------------------------------------
// RFF features + base layernorm -> x[:, 0:259] (f16), zero pad x[:,1027:1056).
__global__ __launch_bounds__(256)
void base_kernel(const float* __restrict__ pos, const float* __restrict__ vals,
                 const float* __restrict__ Bm, const float* __restrict__ g,
                 const float* __restrict__ b, _Float16* __restrict__ x) {
  __shared__ float red[256];
  const int row = blockIdx.x;
  const int tid = threadIdx.x;
  const float px = pos[2 * row], py = pos[2 * row + 1];

  auto feat = [&](int j) -> float {
    if (j < 128) return sinf(px * Bm[j] + py * Bm[128 + j]);
    if (j < 256) return cosf(px * Bm[j - 128] + py * Bm[j]);
    return vals[3 * row + (j - 256)];
  };

  const float v0 = feat(tid);
  const float v1 = (tid < 3) ? feat(256 + tid) : 0.0f;
  const float s  = block_reduce_sum(v0 + v1, red);
  const float sq = block_reduce_sum(v0 * v0 + v1 * v1, red);
  const float mean = s / (float)BASE_IN;
  const float var  = sq / (float)BASE_IN - mean * mean;
  const float rstd = rsqrtf(var + 1e-5f);

  _Float16* xr = x + (size_t)row * XPAD;
  xr[tid] = (_Float16)((v0 - mean) * rstd * g[tid] + b[tid]);
  if (tid < 3)
    xr[256 + tid] = (_Float16)((v1 - mean) * rstd * g[256 + tid] + b[256 + tid]);
  if (tid < XPAD - XDIM) xr[XDIM + tid] = (_Float16)0.0f;
}

// ---------------------------------------------------------------------------
// Brute-force KNN (top-8 nearest, excluding self), LDS-tiled.
__global__ __launch_bounds__(128)
void knn_kernel(const float* __restrict__ pos, int* __restrict__ knn, int n) {
  __shared__ float sx[128], sy[128];
  const int i = blockIdx.x * 128 + threadIdx.x;
  const float px = pos[2 * i], py = pos[2 * i + 1];
  float bd[KNN_K];
  int   bi[KNN_K];
#pragma unroll
  for (int k = 0; k < KNN_K; ++k) { bd[k] = 3.4e38f; bi[k] = 0x7fffffff; }

  for (int t = 0; t < n; t += 128) {
    __syncthreads();
    sx[threadIdx.x] = pos[2 * (t + threadIdx.x)];
    sy[threadIdx.x] = pos[2 * (t + threadIdx.x) + 1];
    __syncthreads();
    for (int j0 = 0; j0 < 128; ++j0) {
      const int j = t + j0;
      if (j == i) continue;
      const float dx = px - sx[j0], dy = py - sy[j0];
      const float d  = dx * dx + dy * dy;
      if (d < bd[KNN_K - 1] || (d == bd[KNN_K - 1] && j < bi[KNN_K - 1])) {
        int p = KNN_K - 1;
        while (p > 0 && (d < bd[p - 1] || (d == bd[p - 1] && j < bi[p - 1]))) {
          bd[p] = bd[p - 1]; bi[p] = bi[p - 1]; --p;
        }
        bd[p] = d; bi[p] = j;
      }
    }
  }
#pragma unroll
  for (int k = 0; k < KNN_K; ++k) knn[i * KNN_K + k] = bi[k];
}

// ---------------------------------------------------------------------------
// Neighbor features (rel pos/dist/vals + LN(6)) and geometric least-squares
// gradients (2x2 Cramer solve + LN(7)), both zero-padded to K=32.
__global__ __launch_bounds__(256)
void nbr_geom_kernel(const float* __restrict__ pos, const float* __restrict__ vals,
                     const int* __restrict__ knn,
                     const float* __restrict__ ng, const float* __restrict__ nb,
                     const float* __restrict__ gg, const float* __restrict__ gb,
                     _Float16* __restrict__ nbr_in,   // (N*K) x 32
                     _Float16* __restrict__ geom_in,  // N x 32
                     int n) {
  const int i = blockIdx.x * blockDim.x + threadIdx.x;
  if (i >= n) return;
  const float px = pos[2 * i], py = pos[2 * i + 1];
  const float vx = vals[3 * i], vy = vals[3 * i + 1], vz = vals[3 * i + 2];
  const float inv2pi = 0.15915494309189535f;

  float a00 = 1e-4f, a01 = 0.0f, a11 = 1e-4f;
  float t0x = 0, t0y = 0, t0z = 0, t1x = 0, t1y = 0, t1z = 0;

  for (int k = 0; k < KNN_K; ++k) {
    const int j = knn[i * KNN_K + k];
    const float rx = (pos[2 * j]     - px) * inv2pi;
    const float ry = (pos[2 * j + 1] - py) * inv2pi;
    const float rd = sqrtf(rx * rx + ry * ry);
    const float ux = vals[3 * j]     - vx;
    const float uy = vals[3 * j + 1] - vy;
    const float uz = vals[3 * j + 2] - vz;
    float raw[6] = {rx, ry, rd, ux, uy, uz};

    float s = 0, sq = 0;
#pragma unroll
    for (int e = 0; e < 6; ++e) { s += raw[e]; sq += raw[e] * raw[e]; }
    const float m = s * (1.0f / 6.0f);
    const float rstd = rsqrtf(sq * (1.0f / 6.0f) - m * m + 1e-5f);
    _Float16* o = nbr_in + (size_t)(i * KNN_K + k) * 32;
#pragma unroll
    for (int e = 0; e < 6; ++e) o[e] = (_Float16)((raw[e] - m) * rstd * ng[e] + nb[e]);
#pragma unroll
    for (int e = 6; e < 32; ++e) o[e] = (_Float16)0.0f;

    a00 += rx * rx; a01 += rx * ry; a11 += ry * ry;
    t0x += rx * ux; t0y += rx * uy; t0z += rx * uz;
    t1x += ry * ux; t1y += ry * uy; t1z += ry * uz;
  }

  const float id = 1.0f / (a00 * a11 - a01 * a01);
  const float s00 = id * ( a11 * t0x - a01 * t1x);  // grad_u[0]
  const float s10 = id * (-a01 * t0x + a00 * t1x);  // grad_u[1]
  const float s01 = id * ( a11 * t0y - a01 * t1y);  // grad_v[0]
  const float s11 = id * (-a01 * t0y + a00 * t1y);  // grad_v[1]
  const float s02 = id * ( a11 * t0z - a01 * t1z);  // grad_p[0]
  const float s12 = id * (-a01 * t0z + a00 * t1z);  // grad_p[1]
  float geom[7] = {s00, s10, s01, s11, s02, s12, s01 - s10};

  float s = 0, sq = 0;
#pragma unroll
  for (int e = 0; e < 7; ++e) { s += geom[e]; sq += geom[e] * geom[e]; }
  const float m = s * (1.0f / 7.0f);
  const float rstd = rsqrtf(sq * (1.0f / 7.0f) - m * m + 1e-5f);
  _Float16* o = geom_in + (size_t)i * 32;
#pragma unroll
  for (int e = 0; e < 7; ++e) o[e] = (_Float16)((geom[e] - m) * rstd * gg[e] + gb[e]);
#pragma unroll
  for (int e = 7; e < 32; ++e) o[e] = (_Float16)0.0f;
}

// ---------------------------------------------------------------------------
// Mean/max over the K=8 neighbor embeddings -> x[:,259:515] / x[:,515:771].
__global__ void nbr_reduce_kernel(const _Float16* __restrict__ emb,
                                  _Float16* __restrict__ x, int n) {
  const int idx = blockIdx.x * blockDim.x + threadIdx.x;
  if (idx >= n * DM) return;
  const int i = idx / DM, d = idx - i * DM;
  const _Float16* e = emb + (size_t)i * KNN_K * DM + d;
  float s = 0.0f, mx = -3.4e38f;
#pragma unroll
  for (int k = 0; k < KNN_K; ++k) {
    const float v = (float)e[k * DM];
    s += v; mx = fmaxf(mx, v);
  }
  _Float16* xr = x + (size_t)i * XPAD;
  xr[BASE_IN + d]      = (_Float16)(s * 0.125f);
  xr[BASE_IN + DM + d] = (_Float16)mx;
}

// ---------------------------------------------------------------------------
// Row layernorm, f16 input -> f16 output (zero-pad [dim, padTo)).
__global__ __launch_bounds__(256)
void ln_f16_kernel(const _Float16* __restrict__ in, int ldi, int dim,
                   const float* __restrict__ g, const float* __restrict__ b,
                   _Float16* __restrict__ out, int ldo, int padTo) {
  __shared__ float red[256];
  const int row = blockIdx.x, tid = threadIdx.x;
  const _Float16* ir = in + (size_t)row * ldi;
  float s = 0, sq = 0;
  for (int j = tid; j < dim; j += 256) { const float v = (float)ir[j]; s += v; sq += v * v; }
  s  = block_reduce_sum(s, red);
  sq = block_reduce_sum(sq, red);
  const float m = s / (float)dim;
  const float rstd = rsqrtf(sq / (float)dim - m * m + 1e-5f);
  _Float16* orow = out + (size_t)row * ldo;
  for (int j = tid; j < dim; j += 256)
    orow[j] = (_Float16)(((float)ir[j] - m) * rstd * g[j] + b[j]);
  for (int j = dim + tid; j < padTo; j += 256) orow[j] = (_Float16)0.0f;
}

// Row layernorm, f32 input -> f16 output.
__global__ __launch_bounds__(256)
void ln_f32_kernel(const float* __restrict__ in, int ldi, int dim,
                   const float* __restrict__ g, const float* __restrict__ b,
                   _Float16* __restrict__ out, int ldo) {
  __shared__ float red[256];
  const int row = blockIdx.x, tid = threadIdx.x;
  const float* ir = in + (size_t)row * ldi;
  float s = 0, sq = 0;
  for (int j = tid; j < dim; j += 256) { const float v = ir[j]; s += v; sq += v * v; }
  s  = block_reduce_sum(s, red);
  sq = block_reduce_sum(sq, red);
  const float m = s / (float)dim;
  const float rstd = rsqrtf(sq / (float)dim - m * m + 1e-5f);
  _Float16* orow = out + (size_t)row * ldo;
  for (int j = tid; j < dim; j += 256)
    orow[j] = (_Float16)((ir[j] - m) * rstd * g[j] + b[j]);
}

// ---------------------------------------------------------------------------
// WMMA GEMM: C[M x N] = epilogue(A[M x K] @ W[K x N] + bias)
// A f16 row-major (M x K, lda = K). Wt f16 row-major TRANSPOSED (N x K,
// ldw = K). Requires M%128==0, N%128==0, K%32==0.
// Block = 256 threads (8 waves); wave (wid&3, wid>>2) owns a 32x64 sub-tile.
// Both LDS tiles are 128 rows x 32 K-halfs; both fragment types load as
// two 8-half (16B) spans at k=kb and 16+kb -> two ds_load_b128 each.
__global__ __launch_bounds__(256)
void gemm_wmma_kernel(const _Float16* __restrict__ A, int lda,
                      const _Float16* __restrict__ Wt, int ldw,
                      const float* __restrict__ bias,
                      _Float16* __restrict__ outH, int ldoH,
                      float* __restrict__ outF, int ldoF,
                      int K, int flags) {
  __shared__ _Float16 ldsA[BM * LDS_STRIDE];
  __shared__ _Float16 ldsB[BN * LDS_STRIDE];

  const int tid  = threadIdx.x;
  const int lane = tid & 31;     // wave32
  const int wid  = tid >> 5;
  const int wm = (wid & 3) * 32;
  const int wn = (wid >> 2) * 64;
  const int blockM = blockIdx.x * BM;
  const int blockN = blockIdx.y * BN;

  v8f acc[2][4] = {};

  // cooperative tile-load coordinates (each thread moves 16 halfs per tile)
  const int trow = tid >> 1;           // 0..127
  const int tcol = (tid & 1) * 16;     // 0 / 16

  // WMMA fragment coordinates (ISA 7.12.2 layouts)
  const int rsel = lane >> 4;          // half-wave select
  const int mrow = lane & 15;          // A row within 16 / B,C column
  const int kb   = rsel * 8;           // A: lanes 16-31 hold K+8 within each 16
  const int koff = rsel * 16;          // unused; B uses kb (same K order as A)

  for (int k0 = 0; k0 < K; k0 += BK) {
    __syncthreads();
    const _Float16* ga = A + (size_t)(blockM + trow) * lda + (k0 + tcol);
    *(half8*)&ldsA[trow * LDS_STRIDE + tcol]     = *(const half8*)(ga);
    *(half8*)&ldsA[trow * LDS_STRIDE + tcol + 8] = *(const half8*)(ga + 8);
    const _Float16* gb = Wt + (size_t)(blockN + trow) * ldw + (k0 + tcol);
    *(half8*)&ldsB[trow * LDS_STRIDE + tcol]     = *(const half8*)(gb);
    *(half8*)&ldsB[trow * LDS_STRIDE + tcol + 8] = *(const half8*)(gb + 8);
    __syncthreads();

    v16h afrag[2];
#pragma unroll
    for (int mt = 0; mt < 2; ++mt) {
      const _Float16* ap = &ldsA[(wm + mt * 16 + mrow) * LDS_STRIDE];
      const half8 lo = *(const half8*)(ap + kb);
      const half8 hi = *(const half8*)(ap + 16 + kb);
#pragma unroll
      for (int e = 0; e < 8; ++e) { afrag[mt][e] = lo[e]; afrag[mt][8 + e] = hi[e]; }
    }
#pragma unroll
    for (int nt = 0; nt < 4; ++nt) {
      // bfrag[e] = Wt[col][kb+e], bfrag[8+e] = Wt[col][16+kb+e] (same K order as A).
      const _Float16* bp = &ldsB[(wn + nt * 16 + mrow) * LDS_STRIDE + kb];
      const half8 blo = *(const half8*)(bp);
      const half8 bhi = *(const half8*)(bp + 16);
      v16h bfrag;
#pragma unroll
      for (int e = 0; e < 8; ++e) { bfrag[e] = blo[e]; bfrag[8 + e] = bhi[e]; }
#pragma unroll
      for (int mt = 0; mt < 2; ++mt)
        acc[mt][nt] = __builtin_amdgcn_wmma_f32_16x16x32_f16(
            false, afrag[mt], false, bfrag, (short)0, acc[mt][nt], false, false);
    }
  }

  // epilogue: bias + optional SiLU, write f16 and/or f32 (+= residual)
  const int rowoff = rsel * 8;
#pragma unroll
  for (int mt = 0; mt < 2; ++mt) {
#pragma unroll
    for (int nt = 0; nt < 4; ++nt) {
      const int gc = blockN + wn + nt * 16 + mrow;
      const float bv = bias ? bias[gc] : 0.0f;
#pragma unroll
      for (int r = 0; r < 8; ++r) {
        const int gr = blockM + wm + mt * 16 + rowoff + r;
        float v = acc[mt][nt][r] + bv;
        if (flags & EPI_SILU) v = v / (1.0f + __expf(-v));
        if (outH) outH[(size_t)gr * ldoH + gc] = (_Float16)v;
        if (outF) {
          if (flags & EPI_RESID) outF[(size_t)gr * ldoF + gc] += v;
          else                   outF[(size_t)gr * ldoF + gc]  = v;
        }
      }
    }
  }
}

// ---------------------------------------------------------------------------
extern "C" void kernel_launch(void* const* d_in, const int* in_sizes, int n_in,
                              void* d_out, int out_size, void* d_ws, size_t ws_size,
                              hipStream_t stream) {
  (void)in_sizes; (void)n_in; (void)out_size; (void)ws_size;

  const float* sensor_vals = (const float*)d_in[0];
  const float* sensor_pos  = (const float*)d_in[1];
  const float* Bmat        = (const float*)d_in[2];
  const float* base_g      = (const float*)d_in[3];
  const float* base_b      = (const float*)d_in[4];

  struct MlpP { const float *ln_g, *ln_b, *w1, *b1, *w2, *b2; };
  auto mlp = [&](int i) -> MlpP {
    return MlpP{(const float*)d_in[i],     (const float*)d_in[i + 1],
                (const float*)d_in[i + 2], (const float*)d_in[i + 3],
                (const float*)d_in[i + 4], (const float*)d_in[i + 5]};
  };
  const MlpP nbrP = mlp(5), geoP = mlp(11), tokP = mlp(17);
  const MlpP blkP[4] = {mlp(23), mlp(29), mlp(35), mlp(41)};
  const MlpP outP = mlp(47);

  // ---- workspace arena (256B-aligned bump allocator) ----
  char* ws = (char*)d_ws;
  size_t off = 0;
  auto alloc = [&](size_t bytes) -> void* {
    void* p = ws + off;
    off += (bytes + 255) & ~(size_t)255;
    return p;
  };

  // transposed f16 weights: Wt is (N x Kp), ld = Kp
  _Float16* w_nbr1 = (_Float16*)alloc((size_t)256 * 32 * 2);
  _Float16* w_nbr2 = (_Float16*)alloc((size_t)256 * 256 * 2);
  _Float16* w_geo1 = (_Float16*)alloc((size_t)256 * 32 * 2);
  _Float16* w_geo2 = (_Float16*)alloc((size_t)256 * 256 * 2);
  _Float16* w_tok1 = (_Float16*)alloc((size_t)HID * XPAD * 2);
  _Float16* w_tok2 = (_Float16*)alloc((size_t)DM * HID * 2);
  _Float16 *w_blk1[4], *w_blk2[4];
  for (int i = 0; i < 4; ++i) {
    w_blk1[i] = (_Float16*)alloc((size_t)HID * DM * 2);
    w_blk2[i] = (_Float16*)alloc((size_t)DM * HID * 2);
  }
  _Float16* w_out1 = (_Float16*)alloc((size_t)HID * DM * 2);
  _Float16* w_out2 = (_Float16*)alloc((size_t)DM * HID * 2);

  int*      knn     = (int*)alloc((size_t)NPTS * KNN_K * 4);
  _Float16* nbr_in  = (_Float16*)alloc((size_t)NPTS * KNN_K * 32 * 2);
  _Float16* geom_in = (_Float16*)alloc((size_t)NPTS * 32 * 2);
  _Float16* x       = (_Float16*)alloc((size_t)NPTS * XPAD * 2);
  float*    tokens  = (float*)alloc((size_t)NPTS * DM * 4);
  _Float16* lnbuf   = (_Float16*)alloc((size_t)NPTS * DM * 2);
  _Float16* hid     = (_Float16*)alloc((size_t)NPTS * HID * 2);
  _Float16* big1    = (_Float16*)alloc((size_t)NPTS * KNN_K * DM * 2); // nbr hidden
  _Float16* big2    = (_Float16*)alloc((size_t)NPTS * KNN_K * DM * 2); // nbr emb
  _Float16* xln     = big2;  // reuse after the neighbor reduce consumes big2

  auto cdiv = [](long a, long b) { return (int)((a + b - 1) / b); };

  // ---- pack weights to f16, transposed, K zero-padded ----
  pack_weight_kernel<<<cdiv(32 * 256, 256), 256, 0, stream>>>(nbrP.w1, w_nbr1, 6, 256, 32);
  pack_weight_kernel<<<cdiv(256 * 256, 256), 256, 0, stream>>>(nbrP.w2, w_nbr2, 256, 256, 256);
  pack_weight_kernel<<<cdiv(32 * 256, 256), 256, 0, stream>>>(geoP.w1, w_geo1, 7, 256, 32);
  pack_weight_kernel<<<cdiv(256 * 256, 256), 256, 0, stream>>>(geoP.w2, w_geo2, 256, 256, 256);
  pack_weight_kernel<<<cdiv((long)XPAD * HID, 256), 256, 0, stream>>>(tokP.w1, w_tok1, XDIM, HID, XPAD);
  pack_weight_kernel<<<cdiv((long)HID * DM, 256), 256, 0, stream>>>(tokP.w2, w_tok2, HID, DM, HID);
  for (int i = 0; i < 4; ++i) {
    pack_weight_kernel<<<cdiv((long)DM * HID, 256), 256, 0, stream>>>(blkP[i].w1, w_blk1[i], DM, HID, DM);
    pack_weight_kernel<<<cdiv((long)HID * DM, 256), 256, 0, stream>>>(blkP[i].w2, w_blk2[i], HID, DM, HID);
  }
  pack_weight_kernel<<<cdiv((long)DM * HID, 256), 256, 0, stream>>>(outP.w1, w_out1, DM, HID, DM);
  pack_weight_kernel<<<cdiv((long)HID * DM, 256), 256, 0, stream>>>(outP.w2, w_out2, HID, DM, HID);

  // ---- features ----
  base_kernel<<<NPTS, 256, 0, stream>>>(sensor_pos, sensor_vals, Bmat, base_g, base_b, x);
  knn_kernel<<<NPTS / 128, 128, 0, stream>>>(sensor_pos, knn, NPTS);
  nbr_geom_kernel<<<NPTS / 256, 256, 0, stream>>>(sensor_pos, sensor_vals, knn,
                                                  nbrP.ln_g, nbrP.ln_b, geoP.ln_g, geoP.ln_b,
                                                  nbr_in, geom_in, NPTS);

  const int MN = NPTS * KNN_K;  // 131072 neighbor rows

  // ---- neighbor MLP (6->256->256) ----
  gemm_wmma_kernel<<<dim3(MN / BM, 256 / BN), 256, 0, stream>>>(
      nbr_in, 32, w_nbr1, 32, nbrP.b1, big1, 256, nullptr, 0, 32, EPI_SILU);
  gemm_wmma_kernel<<<dim3(MN / BM, 256 / BN), 256, 0, stream>>>(
      big1, 256, w_nbr2, 256, nbrP.b2, big2, 256, nullptr, 0, 256, 0);
  nbr_reduce_kernel<<<cdiv((long)NPTS * DM, 256), 256, 0, stream>>>(big2, x, NPTS);

  // ---- geom MLP (7->256->256), second layer lands in x[:,771:1027] ----
  gemm_wmma_kernel<<<dim3(NPTS / BM, 256 / BN), 256, 0, stream>>>(
      geom_in, 32, w_geo1, 32, geoP.b1, hid, 256, nullptr, 0, 32, EPI_SILU);
  gemm_wmma_kernel<<<dim3(NPTS / BM, 256 / BN), 256, 0, stream>>>(
      hid, 256, w_geo2, 256, geoP.b2, x + (BASE_IN + 2 * DM), XPAD, nullptr, 0, 256, 0);

  // ---- token_in (LN(1027) -> 512 silu -> 256) ----
  ln_f16_kernel<<<NPTS, 256, 0, stream>>>(x, XPAD, XDIM, tokP.ln_g, tokP.ln_b, xln, XPAD, XPAD);
  gemm_wmma_kernel<<<dim3(NPTS / BM, HID / BN), 256, 0, stream>>>(
      xln, XPAD, w_tok1, XPAD, tokP.b1, hid, HID, nullptr, 0, XPAD, EPI_SILU);
  gemm_wmma_kernel<<<dim3(NPTS / BM, DM / BN), 256, 0, stream>>>(
      hid, HID, w_tok2, HID, tokP.b2, nullptr, 0, tokens, DM, HID, 0);

  // ---- 4 residual blocks ----
  for (int i = 0; i < 4; ++i) {
    ln_f32_kernel<<<NPTS, 256, 0, stream>>>(tokens, DM, DM, blkP[i].ln_g, blkP[i].ln_b, lnbuf, DM);
    gemm_wmma_kernel<<<dim3(NPTS / BM, HID / BN), 256, 0, stream>>>(
        lnbuf, DM, w_blk1[i], DM, blkP[i].b1, hid, HID, nullptr, 0, DM, EPI_SILU);
    gemm_wmma_kernel<<<dim3(NPTS / BM, DM / BN), 256, 0, stream>>>(
        hid, HID, w_blk2[i], HID, blkP[i].b2, nullptr, 0, tokens, DM, HID, EPI_RESID);
  }

  // ---- out_proj ----
  ln_f32_kernel<<<NPTS, 256, 0, stream>>>(tokens, DM, DM, outP.ln_g, outP.ln_b, lnbuf, DM);
  gemm_wmma_kernel<<<dim3(NPTS / BM, HID / BN), 256, 0, stream>>>(
      lnbuf, DM, w_out1, DM, outP.b1, hid, HID, nullptr, 0, DM, EPI_SILU);
  gemm_wmma_kernel<<<dim3(NPTS / BM, DM / BN), 256, 0, stream>>>(
      hid, HID, w_out2, HID, outP.b2, nullptr, 0, (float*)d_out, DM, HID, 0);
}
